// joint_wise_ffn_32925219291317
// MI455X (gfx1250) — matrix operand order repaired and checked by reading, not compiled, not run on hardware
//
#include <hip/hip_runtime.h>

typedef __attribute__((ext_vector_type(16))) __bf16 v16bf;
typedef __attribute__((ext_vector_type(8)))  __bf16 v8bf;
typedef __attribute__((ext_vector_type(4)))  __bf16 v4bf;
typedef __attribute__((ext_vector_type(8)))  float  v8f;
typedef __attribute__((ext_vector_type(4)))  float  v4f;

// CDNA5 async global->LDS DMA path (ASYNCcnt). Set to 0 to fall back to
// plain global-load + ds-store if the assembler rejects the mnemonic.
#define USE_ASYNC_LDS 1

namespace {
constexpr int Jn = 24;     // joints
constexpr int Dn = 128;    // model dim
constexpr int Hn = 512;    // hidden dim
constexpr int TM = 128;    // token tile per block

// LDS row strides (elements), padded so fragment-load lane stride = 4 banks.
constexpr int WLD = 136;   // W1^T rows: 512 x (128+8) bf16
constexpr int HLD = 520;   // H rows: 128 x (512+8) bf16 ; also W2^T rows
constexpr int XLD = 136;   // X rows: 128 x (128+8) bf16
constexpr int OLD = 132;   // out staging rows: 64 x (128+4) f32

constexpr int W1B = Hn * WLD * 2;   // 139264 B: per-joint W1^T LDS image
constexpr int W2B = Dn * HLD * 2;   // 133120 B: per-joint W2^T LDS image

constexpr int OFF_H = W1B;                        // Hb after Wt region
constexpr int OFF_X = OFF_H + TM * HLD * 2;       // 272384
constexpr int SMEM_BYTES = OFF_X + TM * XLD * 2;  // 307200 (<= 320 KB WGP LDS)
static_assert(SMEM_BYTES <= 320 * 1024, "exceeds WGP LDS");
static_assert(64 * OLD * 4 <= TM * XLD * 2, "out staging must fit X region");
static_assert(W1B % 4096 == 0, "W1 image must be whole 256x16B rounds");
}

// ---------------- async 16B global->LDS copy ----------------
__device__ __forceinline__ void cp16_async(void* lds_dst, unsigned long long gbase,
                                           unsigned voff) {
#if USE_ASYNC_LDS
  const unsigned l =
      (unsigned)(size_t)(__attribute__((address_space(3))) char*)lds_dst;
  asm volatile("global_load_async_to_lds_b128 %0, %1, %2"
               :: "v"(l), "v"(voff), "s"(gbase) : "memory");
#else
  *(v8bf*)lds_dst = *(const v8bf*)(size_t)(gbase + voff);
#endif
}
__device__ __forceinline__ void async_wait0() {
#if USE_ASYNC_LDS
  asm volatile("s_wait_asynccnt 0x0" ::: "memory");
#endif
}

// ---------------- WMMA fragment helpers (ISA 7.12.2 layouts) ----------------
// A (16x32 bf16): lanes 0-15: M=lane, K={0..7,16..23}; lanes 16-31: K={8..15,24..31}
__device__ __forceinline__ v16bf load_a_frag(const __bf16* __restrict__ base,
                                             int row0, int k0, int ld) {
  const int lane = threadIdx.x & 31;
  const int r    = lane & 15;
  const int kh   = (lane >> 4) << 3;
  const __bf16* p = base + (row0 + r) * ld + k0 + kh;
  union { v16bf v; v8bf h[2]; } u;
  u.h[0] = *(const v8bf*)p;
  u.h[1] = *(const v8bf*)(p + 16);
  return u.v;
}
// B (32x16 bf16) from B^T row-major [n][k]: lanes 0-15: N=lane,K={0..7,16..23}; 16-31: K={8..15,24..31}
__device__ __forceinline__ v16bf load_b_frag(const __bf16* __restrict__ baseT,
                                             int n0, int k0, int ld) {
  const int lane = threadIdx.x & 31;
  const int n    = lane & 15;
  const int kh   = (lane >> 4) << 3;
  const __bf16* p = baseT + (n0 + n) * ld + k0 + kh;
  union { v16bf v; v8bf h[2]; } u;
  u.h[0] = *(const v8bf*)p;
  u.h[1] = *(const v8bf*)(p + 16);
  return u.v;
}
__device__ __forceinline__ v8f wmma_bf16(v16bf a, v16bf b, v8f c) {
  return __builtin_amdgcn_wmma_f32_16x16x32_bf16(false, a, false, b,
                                                 (short)0, c, false, false);
}

// ---------------- prep: f32 weights -> bf16 transposed padded LDS images ----------------
__global__ __launch_bounds__(256) void prep_w1(const float* __restrict__ W1,
                                               __bf16* __restrict__ blob) {
  const size_t idx = (size_t)blockIdx.x * 256 + threadIdx.x;  // Jn*Dn*Hn threads
  const int d = (int)(idx & 127);          // fastest -> coalesced blob writes
  const int h = (int)((idx >> 7) & 511);
  const int j = (int)(idx >> 16);
  blob[(size_t)j * (W1B / 2) + h * WLD + d] =
      (__bf16)W1[((size_t)j * Dn + d) * Hn + h];
}
__global__ __launch_bounds__(256) void prep_w2(const float* __restrict__ W2,
                                               __bf16* __restrict__ blob) {
  const size_t idx = (size_t)blockIdx.x * 256 + threadIdx.x;
  const int h = (int)(idx & 511);          // fastest -> coalesced blob writes
  const int d = (int)((idx >> 9) & 127);
  const int j = (int)(idx >> 16);
  blob[(size_t)j * (W2B / 2) + d * HLD + h] =
      (__bf16)W2[((size_t)j * Hn + h) * Dn + d];
}

// ---------------- main fused FFN kernel ----------------
__global__ __launch_bounds__(256) void joint_ffn_kernel(
    const float* __restrict__ x,
    const __bf16* __restrict__ wt1,   // bf16 W1^T LDS images, per joint
    const __bf16* __restrict__ wt2,   // bf16 W2^T LDS images, per joint
    const float* __restrict__ b1, const float* __restrict__ b2,
    float* __restrict__ out) {
  extern __shared__ char smem[];
  __bf16* Wt = (__bf16*)smem;             // phase1: W1^T [512][136]; phase2: W2^T [128][520]
  __bf16* Hb = (__bf16*)(smem + OFF_H);   // hidden tile [128][520] bf16
  __bf16* Xb = (__bf16*)(smem + OFF_X);   // X tile [128][136] bf16 (phase 1 only)
  float*  Ob = (float*)(smem + OFF_X);    // out staging [64][132] f32 (epilogue)

  const int j    = blockIdx.y;
  const int m0   = blockIdx.x * TM;
  const int tid  = threadIdx.x;
  const int wave = tid >> 5;
  const int lane = tid & 31;
  const int cn   = lane & 15;             // C-frag: N = lane%16
  const int cm   = (lane >> 4) << 3;      // C-frag: M offset 0/8
  const int a2     = wave & 3;            // wave's M-pair: rows a2*32 .. a2*32+31
  const int row0   = a2 * 32;
  const int n0base = (wave >> 2) * 4;     // GEMM2 N-tile base (4 tiles/wave)

  // ---- kick off W1^T DMA into LDS (overlapped with X conversion) ----
  {
    const unsigned long long src =
        (unsigned long long)(size_t)(wt1 + (size_t)j * (W1B / 2));
    #pragma unroll 4
    for (int i = 0; i < W1B / 4096; ++i) {              // 34
      const unsigned off = (unsigned)((i * 256 + tid) * 16);
      cp16_async((char*)smem + off, src, off);
    }
  }
  // ---- X tile: coalesced float4 loads, convert to bf16 in LDS ----
  #pragma unroll 4
  for (int i = 0; i < (TM * Dn) / (256 * 4); ++i) {     // 16
    const int idx = i * 256 + tid;
    const int m   = idx >> 5;
    const int d4  = (idx & 31) << 2;
    const v4f v = *(const v4f*)(x + (((size_t)(m0 + m) * Jn) + j) * Dn + d4);
    v4bf b;
    #pragma unroll
    for (int k = 0; k < 4; ++k) b[k] = (__bf16)v[k];
    *(v4bf*)(Xb + m * XLD + d4) = b;
  }
  async_wait0();
  __syncthreads();

  // ---- GEMM1: Hb = relu(X @ W1 + b1); 2 M-tiles x 16 N-tiles per wave ----
  {
    const int nhalf = (wave >> 2) * 16;                 // N-tile base (16 tiles)
    v16bf A[2][4];
    #pragma unroll
    for (int sub = 0; sub < 2; ++sub)
      #pragma unroll
      for (int kt = 0; kt < 4; ++kt)
        A[sub][kt] = load_a_frag(Xb, row0 + sub * 16, kt * 32, XLD);
    const float* b1j = b1 + j * Hn;
    for (int nt = 0; nt < 16; ++nt) {
      const int n0 = (nhalf + nt) * 16;
      v8f c0 = {}, c1 = {};
      #pragma unroll
      for (int kt = 0; kt < 4; ++kt) {
        const v16bf bf = load_b_frag(Wt, n0, kt * 32, WLD);  // reused by both WMMAs
        c0 = wmma_bf16(A[0][kt], bf, c0);
        c1 = wmma_bf16(A[1][kt], bf, c1);
      }
      const float bias = b1j[n0 + cn];
      #pragma unroll
      for (int sub = 0; sub < 2; ++sub) {
        __bf16* hp = Hb + (row0 + sub * 16 + cm) * HLD + n0 + cn;
        #pragma unroll
        for (int r = 0; r < 8; ++r) {
          const float v = (sub ? c1[r] : c0[r]) + bias;
          hp[r * HLD] = (__bf16)(v > 0.f ? v : 0.f);
        }
      }
    }
  }
  __syncthreads();

  // ---- W2^T DMA into LDS (overwrites W1^T region) ----
  {
    const unsigned long long src =
        (unsigned long long)(size_t)(wt2 + (size_t)j * (W2B / 2));
    #pragma unroll 4
    for (int i = 0; i < (W2B + 4095) / 4096; ++i) {     // 33 (tail predicated)
      const unsigned off = (unsigned)((i * 256 + tid) * 16);
      if (off < (unsigned)W2B) cp16_async((char*)smem + off, src, off);
    }
  }
  async_wait0();
  __syncthreads();

  // ---- GEMM2: O = Hb @ W2; 2 M-tiles x 4 N-tiles per wave, K = 512 ----
  v8f c2[2][4] = {};
  for (int kt = 0; kt < Hn / 32; ++kt) {                // 16
    const v16bf a0 = load_a_frag(Hb, row0,      kt * 32, HLD);
    const v16bf a1 = load_a_frag(Hb, row0 + 16, kt * 32, HLD);
    #pragma unroll
    for (int t = 0; t < 4; ++t) {
      const v16bf bf = load_b_frag(Wt, (n0base + t) * 16, kt * 32, HLD);
      c2[0][t] = wmma_bf16(a0, bf, c2[0][t]);
      c2[1][t] = wmma_bf16(a1, bf, c2[1][t]);
    }
  }

  // ---- epilogue: two 64-row halves staged through LDS, bias + coalesced stores ----
  #pragma unroll
  for (int s = 0; s < 2; ++s) {
    if ((a2 >> 1) == s) {                               // waves owning rows s*64..+63
      const int lr0 = (a2 & 1) * 32;
      #pragma unroll
      for (int sub = 0; sub < 2; ++sub)
        #pragma unroll
        for (int t = 0; t < 4; ++t) {
          float* op = Ob + (lr0 + sub * 16 + cm) * OLD + (n0base + t) * 16 + cn;
          #pragma unroll
          for (int r = 0; r < 8; ++r) op[r * OLD] = c2[sub][t][r];
        }
    }
    __syncthreads();
    #pragma unroll
    for (int i = 0; i < (64 * Dn) / (256 * 4); ++i) {   // 8
      const int idx = i * 256 + tid;
      const int m   = idx >> 5;
      const int d4  = (idx & 31) << 2;
      const v4f v  = *(const v4f*)(Ob + m * OLD + d4);
      const v4f bb = *(const v4f*)(b2 + j * Dn + d4);
      *(v4f*)(out + (((size_t)(m0 + s * 64 + m) * Jn) + j) * Dn + d4) = v + bb;
    }
    __syncthreads();
  }
}

extern "C" void kernel_launch(void* const* d_in, const int* in_sizes, int n_in,
                              void* d_out, int out_size, void* d_ws, size_t ws_size,
                              hipStream_t stream) {
  (void)in_sizes; (void)n_in; (void)out_size; (void)ws_size;
  const float* x  = (const float*)d_in[0];
  const float* W1 = (const float*)d_in[1];
  const float* b1 = (const float*)d_in[2];
  const float* W2 = (const float*)d_in[3];
  const float* b2 = (const float*)d_in[4];
  float* out = (float*)d_out;

  __bf16* blob1 = (__bf16*)d_ws;                                   // 24 * 139264 B
  __bf16* blob2 = (__bf16*)((char*)d_ws + (size_t)Jn * W1B);       // 24 * 133120 B

  const int prep_blocks = (Jn * Dn * Hn) / 256;                    // 6144
  prep_w1<<<prep_blocks, 256, 0, stream>>>(W1, blob1);
  prep_w2<<<prep_blocks, 256, 0, stream>>>(W2, blob2);

  (void)hipFuncSetAttribute(reinterpret_cast<const void*>(joint_ffn_kernel),
                            hipFuncAttributeMaxDynamicSharedMemorySize, SMEM_BYTES);
  dim3 grid((32 * 256) / TM, Jn);   // (64, 24)
  joint_ffn_kernel<<<grid, dim3(256), SMEM_BYTES, stream>>>(x, blob1, blob2,
                                                            b1, b2, out);
}
